// TrigramMoE_20641612824629
// MI455X (gfx1250) — matrix operand order repaired and checked by reading, not compiled, not run on hardware
//
#include <hip/hip_runtime.h>
#include <math.h>

// Problem constants (B,T,D,H,E,K) = (2,1024,768,3072,8,2)
#define Bc 2
#define Tc 1024
#define Dc 768
#define Hc 3072
#define Ec 8
#define Nc (Bc * Tc)           // 2048 tokens
#define NEc (Nc * 2)           // 4096 routed entries (top-2)
#define TMc 32                 // M tile (entries)
#define TNc 256                // N tile (cols)
#define KCc 64                 // K chunk staged in LDS per barrier
#define LDK 72                 // padded LDS row stride (bf16 elems), 144B, 16B aligned
#define NEPAD (NEc + Ec * TMc) // 4352: worst case with per-expert TM padding
#define MTILES (NEPAD / TMc)   // 136

typedef __attribute__((ext_vector_type(16))) __bf16 v16bf;
typedef __attribute__((ext_vector_type(8)))  float  v8f;

union ABu { v16bf v; __bf16 e[16]; };
union Cu  { v8f   v; float  e[8]; };

#if defined(__HIP_DEVICE_COMPILE__) && \
    __has_builtin(__builtin_amdgcn_global_load_async_to_lds_b128) && \
    __has_builtin(__builtin_amdgcn_s_wait_asynccnt)
#define USE_ASYNC 1
#define GLOBAL_AS __attribute__((address_space(1)))
#define LDS_AS    __attribute__((address_space(3)))
typedef int v4i_ __attribute__((vector_size(16)));
#else
#define USE_ASYNC 0
#endif

// copy 16B global -> LDS (async path if available)
__device__ __forceinline__ void cp16(__bf16* dst, const __bf16* src) {
#if USE_ASYNC
    __builtin_amdgcn_global_load_async_to_lds_b128(
        (GLOBAL_AS v4i_*)(GLOBAL_AS void*)(void*)src,
        (LDS_AS v4i_*)(LDS_AS void*)(void*)dst, 0, 0);
#else
    *(uint4*)dst = *(const uint4*)src;
#endif
}
__device__ __forceinline__ void stage_fence() {
#if USE_ASYNC
    __builtin_amdgcn_s_wait_asynccnt(0);
#endif
    __syncthreads();
}

// ---------------------------------------------------------------------------
// init: entry_token = -1 (padding marker), counts = 0
__global__ void init_kernel(int* entry_token, int* counts) {
    int i = blockIdx.x * blockDim.x + threadIdx.x;
    if (i < NEPAD) entry_token[i] = -1;
    if (i < Ec) counts[i] = 0;
}

// ---------------------------------------------------------------------------
// routing: one wave32 per token.
__global__ void route_kernel(const float* __restrict__ x,
                             const float* __restrict__ Wr,
                             float* __restrict__ gates,
                             int* __restrict__ topidx,
                             float* __restrict__ probsN,
                             int* __restrict__ counts) {
    int gid  = blockIdx.x * blockDim.x + threadIdx.x;
    int tok  = gid >> 5;
    int lane = gid & 31;
    if (tok >= Nc) return;
    const float* xr = x + (size_t)tok * Dc;
    float z0 = 0.f, z1 = 0.f, z2 = 0.f;
    for (int d = lane; d < Dc; d += 32) {
        float xv = xr[d];
        z0 = fmaf(xv, Wr[d], z0);
        z1 = fmaf(xv, Wr[Dc + d], z1);
        z2 = fmaf(xv, Wr[2 * Dc + d], z2);
    }
#pragma unroll
    for (int o = 16; o >= 1; o >>= 1) {
        z0 += __shfl_xor(z0, o, 32);
        z1 += __shfl_xor(z1, o, 32);
        z2 += __shfl_xor(z2, o, 32);
    }
    if (lane == 0) {
        const float inv = 0.57735026918962576f; // 1/sqrt(3)
        float l[Ec];
        float lmax = -3.402823e38f;
#pragma unroll
        for (int e = 0; e < Ec; ++e) {
            float a = (e & 4) ? z0 : -z0;
            float b = (e & 2) ? z1 : -z1;
            float c = (e & 1) ? z2 : -z2;
            l[e] = (a + b + c) * inv;
            lmax = fmaxf(lmax, l[e]);
        }
        float p[Ec], s = 0.f;
#pragma unroll
        for (int e = 0; e < Ec; ++e) { p[e] = expf(l[e] - lmax); s += p[e]; }
        float isum = 1.0f / s;
#pragma unroll
        for (int e = 0; e < Ec; ++e) { p[e] *= isum; probsN[tok * Ec + e] = p[e]; }
        int i1 = 0;
#pragma unroll
        for (int e = 1; e < Ec; ++e) if (p[e] > p[i1]) i1 = e;
        int i2 = (i1 == 0) ? 1 : 0;
#pragma unroll
        for (int e = 0; e < Ec; ++e) if (e != i1 && p[e] > p[i2]) i2 = e;
        float g = p[i1] + p[i2];
        gates[tok * 2 + 0] = p[i1] / g;
        gates[tok * 2 + 1] = p[i2] / g;
        topidx[tok * 2 + 0] = i1;
        topidx[tok * 2 + 1] = i2;
        atomicAdd(&counts[i1], 1);
        atomicAdd(&counts[i2], 1);
    }
}

// deterministic per-expert prob sums (fixed-order tree reduction)
__global__ void probreduce_kernel(const float* __restrict__ probsN,
                                  float* __restrict__ probsum) {
    __shared__ float s[256];
    int e = blockIdx.x;
    float acc = 0.f;
    for (int n = threadIdx.x; n < Nc; n += 256) acc += probsN[n * Ec + e];
    s[threadIdx.x] = acc;
    __syncthreads();
    for (int st = 128; st > 0; st >>= 1) {
        if ((int)threadIdx.x < st) s[threadIdx.x] += s[threadIdx.x + st];
        __syncthreads();
    }
    if (threadIdx.x == 0) probsum[e] = s[0];
}

// padded segment offsets + scatter cursors
__global__ void offsets_kernel(const int* __restrict__ counts,
                               int* __restrict__ offs, int* __restrict__ cursor) {
    if (threadIdx.x == 0 && blockIdx.x == 0) {
        int acc = 0;
        for (int e = 0; e < Ec; ++e) {
            offs[e] = acc;
            cursor[e] = acc;
            acc += ((counts[e] + TMc - 1) / TMc) * TMc;
        }
        offs[Ec] = acc;
    }
}

// scatter tokens into expert-contiguous entry slots
__global__ void scatter_kernel(const int* __restrict__ topidx,
                               int* __restrict__ cursor,
                               int* __restrict__ entry_token,
                               int* __restrict__ token_slot) {
    int n = blockIdx.x * blockDim.x + threadIdx.x;
    if (n >= Nc) return;
#pragma unroll
    for (int k = 0; k < 2; ++k) {
        int e = topidx[n * 2 + k];
        int slot = atomicAdd(&cursor[e], 1);
        entry_token[slot] = n;
        token_slot[n * 2 + k] = slot;
    }
}

// x fp32 -> bf16
__global__ void convx_kernel(const float* __restrict__ x, __bf16* __restrict__ xbf) {
    int i = blockIdx.x * blockDim.x + threadIdx.x;
    if (i < Nc * Dc) xbf[i] = (__bf16)x[i];
}

// ---------------------------------------------------------------------------
// W (E, R, C) fp32 -> Wt (E, C, R) bf16  (tiled LDS transpose, coalesced both sides)
__global__ __launch_bounds__(256) void transpose_kernel(
        const float* __restrict__ W, __bf16* __restrict__ Wt, int R, int C) {
    __shared__ float tile[32][33];
    const float* We = W + (size_t)blockIdx.z * R * C;
    __bf16* Wte = Wt + (size_t)blockIdx.z * R * C;
    int c0 = blockIdx.x * 32, r0 = blockIdx.y * 32;
    int tx = threadIdx.x & 31, ty = threadIdx.x >> 5; // ty in 0..7
#pragma unroll
    for (int i = 0; i < 32; i += 8)
        tile[ty + i][tx] = We[(size_t)(r0 + ty + i) * C + c0 + tx];
    __syncthreads();
#pragma unroll
    for (int i = 0; i < 32; i += 8)
        Wte[(size_t)(c0 + ty + i) * R + r0 + tx] = (__bf16)tile[tx][ty + i];
}

// ---------------------------------------------------------------------------
// GEMM1: h[entry, H] = gelu( Xg[entry, D] @ W1[e][D, H] ).
// B source is pre-transposed bf16 W1t[e][H][D] -> rows are K-contiguous, so both
// A and B tiles stage with linear (async) 16B copies. KC=64 per barrier,
// 2 WMMA micro-steps x 4 N-frags per wave per stage.
__global__ __launch_bounds__(256) void gemm1_kernel(
        const __bf16* __restrict__ xbf, const __bf16* __restrict__ W1t,
        const int* __restrict__ offs, const int* __restrict__ entry_token,
        __bf16* __restrict__ hbf) {
    __shared__ __align__(16) __bf16 sA[TMc][LDK];
    __shared__ __align__(16) __bf16 sB[TNc][LDK];
    __shared__ int s_tok[TMc];
    __shared__ int s_e;
    const int m0 = blockIdx.y * TMc;
    const int h0 = blockIdx.x * TNc;
    if (threadIdx.x == 0) {
        int e = -1;
        if (m0 < offs[Ec])
            for (int i = 0; i < Ec; ++i)
                if (m0 >= offs[i] && m0 < offs[i + 1]) e = i;
        s_e = e;
    }
    if ((int)threadIdx.x < TMc) s_tok[threadIdx.x] = entry_token[m0 + threadIdx.x];
    __syncthreads();
    const int e = s_e;
    if (e < 0) return;

    const int wave = threadIdx.x >> 5, lane = threadIdx.x & 31;
    const int wm = wave & 1, wn = wave >> 1;
    const int l16 = lane & 15, khalf = lane >> 4;
    const int ar = threadIdx.x >> 3, ac = (threadIdx.x & 7) * 8; // A-stage row/col
    const int atok = s_tok[ar];

    Cu acc[4];
#pragma unroll
    for (int f = 0; f < 4; ++f)
#pragma unroll
        for (int r = 0; r < 8; ++r) acc[f].e[r] = 0.f;

    const __bf16* W1te = W1t + (size_t)e * Hc * Dc;
    const __bf16* brow = W1te + (size_t)(h0 + threadIdx.x) * Dc;

    for (int k0 = 0; k0 < Dc; k0 += KCc) {
        // stage A: 32 rows x 64 k, one 16B chunk per thread
        if (atok >= 0) {
            cp16(&sA[ar][ac], xbf + (size_t)atok * Dc + k0 + ac);
        } else {
            uint4 z = {0u, 0u, 0u, 0u};
            *(uint4*)&sA[ar][ac] = z;
        }
        // stage B: thread t owns LDS row t (one output col), 8 x 16B chunks
        __builtin_prefetch(brow + k0 + KCc, 0, 0);
#pragma unroll
        for (int c = 0; c < 8; ++c)
            cp16(&sB[threadIdx.x][c * 8], brow + k0 + c * 8);
        stage_fence();

#pragma unroll
        for (int ks = 0; ks < KCc; ks += 32) {
            ABu a;
            const int mrow = wm * 16 + l16;
#pragma unroll
            for (int i = 0; i < 8; ++i) a.e[i] = sA[mrow][ks + khalf * 8 + i];
#pragma unroll
            for (int i = 0; i < 8; ++i) a.e[8 + i] = sA[mrow][ks + 16 + khalf * 8 + i];
#pragma unroll
            for (int f = 0; f < 4; ++f) {
                ABu b;
                const int ncol = wn * 64 + f * 16 + l16;
#pragma unroll
                for (int i = 0; i < 16; ++i) b.e[i] = sB[ncol][ks + (i >> 3) * 16 + khalf * 8 + (i & 7)];
                acc[f].v = __builtin_amdgcn_wmma_f32_16x16x32_bf16(
                    false, a.v, false, b.v, (short)0, acc[f].v, false, false);
            }
        }
        __syncthreads();
    }
    // epilogue: exact GELU, store bf16
#pragma unroll
    for (int f = 0; f < 4; ++f) {
        const int ncol = h0 + wn * 64 + f * 16 + l16;
#pragma unroll
        for (int r = 0; r < 8; ++r) {
            const int m = m0 + wm * 16 + r + 8 * khalf;
            float v = acc[f].e[r];
            v = 0.5f * v * (1.0f + erff(v * 0.70710678118654752f));
            hbf[(size_t)m * Hc + ncol] = (__bf16)v;
        }
    }
}

// ---------------------------------------------------------------------------
// GEMM2: y[entry, D] = h[entry, H] @ W2[e][H, D]; B from W2t[e][D][H].
__global__ __launch_bounds__(256) void gemm2_kernel(
        const __bf16* __restrict__ hbf, const __bf16* __restrict__ W2t,
        const int* __restrict__ offs, float* __restrict__ ybuf) {
    __shared__ __align__(16) __bf16 sA[TMc][LDK];
    __shared__ __align__(16) __bf16 sB[TNc][LDK];
    __shared__ int s_e;
    const int m0 = blockIdx.y * TMc;
    const int d0 = blockIdx.x * TNc;
    if (threadIdx.x == 0) {
        int e = -1;
        if (m0 < offs[Ec])
            for (int i = 0; i < Ec; ++i)
                if (m0 >= offs[i] && m0 < offs[i + 1]) e = i;
        s_e = e;
    }
    __syncthreads();
    const int e = s_e;
    if (e < 0) return;

    const int wave = threadIdx.x >> 5, lane = threadIdx.x & 31;
    const int wm = wave & 1, wn = wave >> 1;
    const int l16 = lane & 15, khalf = lane >> 4;
    const int ar = threadIdx.x >> 3, ac = (threadIdx.x & 7) * 8;

    Cu acc[4];
#pragma unroll
    for (int f = 0; f < 4; ++f)
#pragma unroll
        for (int r = 0; r < 8; ++r) acc[f].e[r] = 0.f;

    const __bf16* W2te = W2t + (size_t)e * Dc * Hc;
    const __bf16* brow = W2te + (size_t)(d0 + threadIdx.x) * Hc;
    const __bf16* arow = hbf + (size_t)(m0 + ar) * Hc;

    for (int k0 = 0; k0 < Hc; k0 += KCc) {
        cp16(&sA[ar][ac], arow + k0 + ac);
        __builtin_prefetch(brow + k0 + KCc, 0, 0);
#pragma unroll
        for (int c = 0; c < 8; ++c)
            cp16(&sB[threadIdx.x][c * 8], brow + k0 + c * 8);
        stage_fence();

#pragma unroll
        for (int ks = 0; ks < KCc; ks += 32) {
            ABu a;
            const int mrow = wm * 16 + l16;
#pragma unroll
            for (int i = 0; i < 8; ++i) a.e[i] = sA[mrow][ks + khalf * 8 + i];
#pragma unroll
            for (int i = 0; i < 8; ++i) a.e[8 + i] = sA[mrow][ks + 16 + khalf * 8 + i];
#pragma unroll
            for (int f = 0; f < 4; ++f) {
                ABu b;
                const int ncol = wn * 64 + f * 16 + l16;
#pragma unroll
                for (int i = 0; i < 16; ++i) b.e[i] = sB[ncol][ks + (i >> 3) * 16 + khalf * 8 + (i & 7)];
                acc[f].v = __builtin_amdgcn_wmma_f32_16x16x32_bf16(
                    false, a.v, false, b.v, (short)0, acc[f].v, false, false);
            }
        }
        __syncthreads();
    }
#pragma unroll
    for (int f = 0; f < 4; ++f) {
        const int ncol = d0 + wn * 64 + f * 16 + l16;
#pragma unroll
        for (int r = 0; r < 8; ++r) {
            const int m = m0 + wm * 16 + r + 8 * khalf;
            ybuf[(size_t)m * Dc + ncol] = acc[f].e[r];
        }
    }
}

// ---------------------------------------------------------------------------
// gather: out[n,:] = g0 * y[slot0,:] + g1 * y[slot1,:]  (2-term add: bitwise det.)
__global__ void gather_kernel(const float* __restrict__ ybuf,
                              const float* __restrict__ gates,
                              const int* __restrict__ token_slot,
                              float* __restrict__ out) {
    int i = blockIdx.x * blockDim.x + threadIdx.x;
    if (i >= Nc * Dc) return;
    int n = i / Dc, d = i - n * Dc;
    int s0 = token_slot[n * 2 + 0], s1 = token_slot[n * 2 + 1];
    out[i] = gates[n * 2 + 0] * ybuf[(size_t)s0 * Dc + d] +
             gates[n * 2 + 1] * ybuf[(size_t)s1 * Dc + d];
}

__global__ void aux_kernel(const float* __restrict__ probsum, float* __restrict__ outaux) {
    if (threadIdx.x == 0 && blockIdx.x == 0) {
        float s = 0.f;
        for (int e = 0; e < Ec; ++e) {
            float dlt = probsum[e] / (float)Nc - 1.0f / (float)Ec;
            s += dlt * dlt;
        }
        outaux[0] = 0.01f * (s / (float)Ec);
    }
}

// ---------------------------------------------------------------------------
extern "C" void kernel_launch(void* const* d_in, const int* in_sizes, int n_in,
                              void* d_out, int out_size, void* d_ws, size_t ws_size,
                              hipStream_t stream) {
    (void)in_sizes; (void)n_in; (void)out_size; (void)ws_size;
    const float* x  = (const float*)d_in[0];   // (B,T,D)
    const float* Wr = (const float*)d_in[1];   // (3,D)
    const float* W1 = (const float*)d_in[2];   // (E,D,H)
    const float* W2 = (const float*)d_in[3];   // (E,H,D)
    float* out = (float*)d_out;                // N*D floats + 1 aux float

    char* base = (char*)d_ws;
    size_t off = 0;
    auto carve = [&](size_t bytes) -> char* {
        char* p = base + off;
        off = (off + bytes + 255) & ~(size_t)255;
        return p;
    };
    float* probsum     = (float*)carve(Ec * sizeof(float));
    int*   counts      = (int*)carve(Ec * sizeof(int));
    int*   offs        = (int*)carve((Ec + 1) * sizeof(int));
    int*   cursor      = (int*)carve(Ec * sizeof(int));
    float* gates       = (float*)carve((size_t)Nc * 2 * sizeof(float));
    int*   topidx      = (int*)carve((size_t)Nc * 2 * sizeof(int));
    int*   entry_token = (int*)carve((size_t)NEPAD * sizeof(int));
    int*   token_slot  = (int*)carve((size_t)Nc * 2 * sizeof(int));
    float* probsN      = (float*)carve((size_t)Nc * Ec * sizeof(float));
    __bf16* xbf        = (__bf16*)carve((size_t)Nc * Dc * sizeof(__bf16));
    __bf16* w1t        = (__bf16*)carve((size_t)Ec * Dc * Hc * sizeof(__bf16));
    __bf16* w2t        = (__bf16*)carve((size_t)Ec * Hc * Dc * sizeof(__bf16));
    __bf16* hbf        = (__bf16*)carve((size_t)NEPAD * Hc * sizeof(__bf16));
    float*  ybuf       = (float*)carve((size_t)NEPAD * Dc * sizeof(float));

    init_kernel<<<(NEPAD + 255) / 256, 256, 0, stream>>>(entry_token, counts);
    route_kernel<<<(Nc * 32) / 128, 128, 0, stream>>>(x, Wr, gates, topidx, probsN, counts);
    probreduce_kernel<<<Ec, 256, 0, stream>>>(probsN, probsum);
    offsets_kernel<<<1, 32, 0, stream>>>(counts, offs, cursor);
    scatter_kernel<<<(Nc + 255) / 256, 256, 0, stream>>>(topidx, cursor, entry_token, token_slot);
    convx_kernel<<<(Nc * Dc + 255) / 256, 256, 0, stream>>>(x, xbf);
    // W1 (E,D,H) -> W1t (E,H,D);  W2 (E,H,D) -> W2t (E,D,H)
    transpose_kernel<<<dim3(Hc / 32, Dc / 32, Ec), 256, 0, stream>>>(W1, w1t, Dc, Hc);
    transpose_kernel<<<dim3(Dc / 32, Hc / 32, Ec), 256, 0, stream>>>(W2, w2t, Hc, Dc);
    gemm1_kernel<<<dim3(Hc / TNc, MTILES), 256, 0, stream>>>(xbf, w1t, offs, entry_token, hbf);
    gemm2_kernel<<<dim3(Dc / TNc, MTILES), 256, 0, stream>>>(hbf, w2t, offs, ybuf);
    gather_kernel<<<(Nc * Dc + 255) / 256, 256, 0, stream>>>(ybuf, gates, token_slot, out);
    aux_kernel<<<1, 32, 0, stream>>>(probsum, out + (size_t)Nc * Dc);
}
